// MultiHeadAttention_40810779246812
// MI455X (gfx1250) — matrix operand order repaired and checked by reading, compiled, not run on hardware
//
#include <hip/hip_runtime.h>

#define DIM 768
#define HEADS 12
#define HEAD_DIM 64
#define BATCH 8
#define SEQ 1024
#define MROWS (BATCH * SEQ)   // 8192
#define QKV_N (3 * DIM)       // 2304
#define SOFT_SCALE 0.125f     // 64^-0.5

#define TM 128
#define TN 64
#define TK 64
#define LDA 72                // LDS row stride in bf16 (144B, 16B-aligned rows)

typedef __bf16 bf16_t;
typedef __attribute__((ext_vector_type(16))) __bf16 v16bf;
typedef __attribute__((ext_vector_type(8)))  __bf16 v8bf;
typedef __attribute__((ext_vector_type(8)))  float  v8f;

__device__ __forceinline__ v8f wmma_bf16(v16bf a, v16bf b, v8f c) {
  // D(16x16 f32) = A(16x32 bf16) * B(32x16 bf16) + C
  return __builtin_amdgcn_wmma_f32_16x16x32_bf16(false, a, false, b, (short)0, c,
                                                 false, false);
}

// A-layout fragment: lane holds row M=lr; elems 0..7 at K=half*8.., 8..15 at K=16+half*8..
__device__ __forceinline__ v16bf frag_a(const bf16_t* lo, const bf16_t* hi) {
  v8bf x = *(const v8bf*)lo;
  v8bf y = *(const v8bf*)hi;
  v16bf r;
#pragma unroll
  for (int i = 0; i < 8; ++i) { r[i] = x[i]; r[i + 8] = y[i]; }
  return r;
}

// B-layout fragment: lane holds column N=lr; elems 0..7 at K=half*8.., 8..15 at K=16+half*8..
__device__ __forceinline__ v16bf frag_b(const bf16_t* p) {
  return frag_a(p, p + 16);
}

__device__ __forceinline__ void cvt_store8(bf16_t* dst, const float* src) {
  float4 a = *(const float4*)src;
  float4 b = *(const float4*)(src + 4);
  dst[0] = (bf16_t)a.x; dst[1] = (bf16_t)a.y;
  dst[2] = (bf16_t)a.z; dst[3] = (bf16_t)a.w;
  dst[4] = (bf16_t)b.x; dst[5] = (bf16_t)b.y;
  dst[6] = (bf16_t)b.z; dst[7] = (bf16_t)b.w;
}

// ---------------------------------------------------------------------------
// Kernel 1: qkv = x @ w_qkv^T, scatter to Q[B,H,N,64], K[B,H,N,64], Vt[B,H,64,N]
// Tile invariants: N-tile (64-aligned, 64 wide) stays inside one of q/k/v and
// one head; M-tile (128-aligned, 128 tall) stays inside one batch row block.
// ---------------------------------------------------------------------------
__global__ __launch_bounds__(256) void qkv_gemm_kernel(
    const float* __restrict__ x,      // [8192, 768]
    const float* __restrict__ w,      // [2304, 768]
    bf16_t* __restrict__ qw, bf16_t* __restrict__ kw, bf16_t* __restrict__ vtw) {
  __shared__ __align__(16) bf16_t As[TM][LDA];
  __shared__ __align__(16) bf16_t Bs[TN][LDA];
  const int n0 = blockIdx.x * TN;
  const int m0 = blockIdx.y * TM;
  const int t = threadIdx.x;
  const int lane = t & 31, wv = t >> 5;
  const int lr = lane & 15, half = lane >> 4;
  const int wm = wv >> 1, wn = wv & 1;   // 4x2 wave grid, each wave 32x32

  v8f acc[2][2] = {};
  for (int k0 = 0; k0 < DIM; k0 += TK) {
    {  // A: 128x64 f32 -> bf16 LDS (32 elems/thread)
      const int row = t >> 1, cb = (t & 1) * 32;
      const float* src = x + (size_t)(m0 + row) * DIM + k0 + cb;
      __builtin_prefetch(src + TK, 0, 3);
#pragma unroll
      for (int i = 0; i < 32; i += 8) cvt_store8(&As[row][cb + i], src + i);
    }
    {  // B: 64 rows of W, 64 K each (16 elems/thread)
      const int row = t >> 2, cb = (t & 3) * 16;
      const float* src = w + (size_t)(n0 + row) * DIM + k0 + cb;
      __builtin_prefetch(src + TK, 0, 3);
      cvt_store8(&Bs[row][cb], src);
      cvt_store8(&Bs[row][cb + 8], src + 8);
    }
    __syncthreads();
    v16bf afrag[2][2], bfrag[2][2];
#pragma unroll
    for (int mt = 0; mt < 2; ++mt) {
      const int r = wm * 32 + mt * 16 + lr;
#pragma unroll
      for (int s = 0; s < 2; ++s)
        afrag[mt][s] = frag_a(&As[r][s * 32 + half * 8], &As[r][s * 32 + half * 8 + 16]);
    }
#pragma unroll
    for (int nt = 0; nt < 2; ++nt) {
      const int r = wn * 32 + nt * 16 + lr;
#pragma unroll
      for (int s = 0; s < 2; ++s)
        bfrag[nt][s] = frag_b(&Bs[r][s * 32 + half * 8]);
    }
#pragma unroll
    for (int mt = 0; mt < 2; ++mt)
#pragma unroll
      for (int nt = 0; nt < 2; ++nt)
#pragma unroll
        for (int s = 0; s < 2; ++s)
          acc[mt][nt] = wmma_bf16(afrag[mt][s], bfrag[nt][s], acc[mt][nt]);
    __syncthreads();
  }
  // Epilogue scatter: all selector math is block-uniform (scalar).
  const int which = n0 / DIM;          // 0=q, 1=k, 2=v
  const int h = (n0 % DIM) >> 6;       // head index
  const int b = m0 >> 10;              // batch index
  const int nbase = m0 & 1023;
  const size_t bhofs = (size_t)(b * HEADS + h);
  if (which < 2) {
    bf16_t* dst = (which == 0 ? qw : kw) + bhofs * SEQ * HEAD_DIM;
#pragma unroll
    for (int mt = 0; mt < 2; ++mt)
#pragma unroll
      for (int nt = 0; nt < 2; ++nt) {
        const int dh = wn * 32 + nt * 16 + lr;
#pragma unroll
        for (int r = 0; r < 8; ++r) {
          const int n = nbase + wm * 32 + mt * 16 + r + half * 8;
          dst[(size_t)n * HEAD_DIM + dh] = (bf16_t)acc[mt][nt][r];
        }
      }
  } else {
    bf16_t* dst = vtw + bhofs * HEAD_DIM * SEQ;
#pragma unroll
    for (int mt = 0; mt < 2; ++mt)
#pragma unroll
      for (int nt = 0; nt < 2; ++nt) {
        const int dh = wn * 32 + nt * 16 + lr;
#pragma unroll
        for (int r = 0; r < 8; ++r) {
          const int n = nbase + wm * 32 + mt * 16 + r + half * 8;
          dst[(size_t)dh * SEQ + n] = (bf16_t)acc[mt][nt][r];
        }
      }
  }
}

// ---------------------------------------------------------------------------
// Kernel 2: flash attention, 1 WG per (bh, 64-query block), 4 waves x 16 rows
// ---------------------------------------------------------------------------
__global__ __launch_bounds__(128) void attn_kernel(
    const bf16_t* __restrict__ qw, const bf16_t* __restrict__ kw,
    const bf16_t* __restrict__ vtw, bf16_t* __restrict__ ow /* [8192,768] */) {
  __shared__ __align__(16) bf16_t Pl[4][16][72];
  const int bh = blockIdx.y;
  const int qblk = blockIdx.x;
  const int t = threadIdx.x;
  const int lane = t & 31, wv = t >> 5;
  const int lr = lane & 15, half = lane >> 4;

  const bf16_t* qp = qw + (size_t)bh * SEQ * HEAD_DIM;
  const bf16_t* kp = kw + (size_t)bh * SEQ * HEAD_DIM;
  const bf16_t* vp = vtw + (size_t)bh * HEAD_DIM * SEQ;
  const int qr0 = qblk * 64 + wv * 16;

  v16bf qfrag[2];
#pragma unroll
  for (int s = 0; s < 2; ++s) {
    const bf16_t* base = qp + (size_t)(qr0 + lr) * HEAD_DIM + s * 32 + half * 8;
    qfrag[s] = frag_a(base, base + 16);
  }

  v8f o[4] = {};
  float mstat[8], lstat[8];
#pragma unroll
  for (int r = 0; r < 8; ++r) { mstat[r] = -1e30f; lstat[r] = 0.0f; }

  for (int j = 0; j < SEQ / 64; ++j) {
    v8f st[4] = {};
#pragma unroll
    for (int nt = 0; nt < 4; ++nt) {
      const bf16_t* kb = kp + (size_t)(j * 64 + nt * 16 + lr) * HEAD_DIM + half * 8;
#pragma unroll
      for (int s = 0; s < 2; ++s)
        st[nt] = wmma_bf16(qfrag[s], frag_b(kb + s * 32), st[nt]);
    }
    // Online softmax: row (r + 8*half) lives in VGPR r across a 16-lane half
#pragma unroll
    for (int r = 0; r < 8; ++r) {
      float mx = -1e30f;
#pragma unroll
      for (int nt = 0; nt < 4; ++nt) {
        st[nt][r] *= SOFT_SCALE;
        mx = fmaxf(mx, st[nt][r]);
      }
#pragma unroll
      for (int off = 1; off < 16; off <<= 1) mx = fmaxf(mx, __shfl_xor(mx, off, 32));
      const float newm = fmaxf(mstat[r], mx);
      const float corr = __expf(mstat[r] - newm);
      mstat[r] = newm;
      float rs = 0.0f;
#pragma unroll
      for (int nt = 0; nt < 4; ++nt) {
        const float p = __expf(st[nt][r] - newm);
        st[nt][r] = p;
        rs += p;
      }
#pragma unroll
      for (int off = 1; off < 16; off <<= 1) rs += __shfl_xor(rs, off, 32);
      lstat[r] = lstat[r] * corr + rs;
#pragma unroll
      for (int ot = 0; ot < 4; ++ot) o[ot][r] *= corr;
    }
    // C layout -> A layout via per-wave LDS tile (wave-internal: dscnt wait only)
#pragma unroll
    for (int nt = 0; nt < 4; ++nt)
#pragma unroll
      for (int r = 0; r < 8; ++r)
        Pl[wv][r + half * 8][nt * 16 + lr] = (bf16_t)st[nt][r];
    asm volatile("s_wait_dscnt 0" ::: "memory");
    v16bf pfrag[2];
#pragma unroll
    for (int s = 0; s < 2; ++s)
      pfrag[s] = frag_a(&Pl[wv][lr][s * 32 + half * 8],
                        &Pl[wv][lr][s * 32 + half * 8 + 16]);
#pragma unroll
    for (int ot = 0; ot < 4; ++ot) {
      const bf16_t* vb = vp + (size_t)(ot * 16 + lr) * SEQ + j * 64 + half * 8;
#pragma unroll
      for (int s = 0; s < 2; ++s)
        o[ot] = wmma_bf16(pfrag[s], frag_b(vb + s * 32), o[ot]);
    }
  }
  // Normalize and write o as bf16 [B*N, 768] with head interleave restored
  const int b = bh / HEADS, h = bh % HEADS;
#pragma unroll
  for (int r = 0; r < 8; ++r) {
    const float inv = 1.0f / lstat[r];
    const int nq = qr0 + r + half * 8;
    const size_t gm = (size_t)b * SEQ + nq;
#pragma unroll
    for (int ot = 0; ot < 4; ++ot) {
      const int col = h * 64 + ot * 16 + lr;
      ow[gm * DIM + col] = (bf16_t)(o[ot][r] * inv);
    }
  }
}

// ---------------------------------------------------------------------------
// Kernel 3: out = o @ w_out^T + b_out  (f32 result)
// ---------------------------------------------------------------------------
__global__ __launch_bounds__(256) void out_gemm_kernel(
    const bf16_t* __restrict__ ob,   // [8192, 768] bf16
    const float* __restrict__ w,     // [768, 768]
    const float* __restrict__ bias,  // [768]
    float* __restrict__ out) {       // [8192, 768]
  __shared__ __align__(16) bf16_t As[TM][LDA];
  __shared__ __align__(16) bf16_t Bs[TN][LDA];
  const int n0 = blockIdx.x * TN;
  const int m0 = blockIdx.y * TM;
  const int t = threadIdx.x;
  const int lane = t & 31, wv = t >> 5;
  const int lr = lane & 15, half = lane >> 4;
  const int wm = wv >> 1, wn = wv & 1;

  v8f acc[2][2] = {};
  for (int k0 = 0; k0 < DIM; k0 += TK) {
    {  // A already bf16: 128x64 (32 elems/thread)
      const int row = t >> 1, cb = (t & 1) * 32;
      const bf16_t* src = ob + (size_t)(m0 + row) * DIM + k0 + cb;
      __builtin_prefetch(src + TK, 0, 3);
#pragma unroll
      for (int i = 0; i < 32; i += 8) *(v8bf*)&As[row][cb + i] = *(const v8bf*)(src + i);
    }
    {  // B: f32 -> bf16 (16 elems/thread)
      const int row = t >> 2, cb = (t & 3) * 16;
      const float* src = w + (size_t)(n0 + row) * DIM + k0 + cb;
      __builtin_prefetch(src + TK, 0, 3);
      cvt_store8(&Bs[row][cb], src);
      cvt_store8(&Bs[row][cb + 8], src + 8);
    }
    __syncthreads();
    v16bf afrag[2][2], bfrag[2][2];
#pragma unroll
    for (int mt = 0; mt < 2; ++mt) {
      const int r = wm * 32 + mt * 16 + lr;
#pragma unroll
      for (int s = 0; s < 2; ++s)
        afrag[mt][s] = frag_a(&As[r][s * 32 + half * 8], &As[r][s * 32 + half * 8 + 16]);
    }
#pragma unroll
    for (int nt = 0; nt < 2; ++nt) {
      const int r = wn * 32 + nt * 16 + lr;
#pragma unroll
      for (int s = 0; s < 2; ++s)
        bfrag[nt][s] = frag_b(&Bs[r][s * 32 + half * 8]);
    }
#pragma unroll
    for (int mt = 0; mt < 2; ++mt)
#pragma unroll
      for (int nt = 0; nt < 2; ++nt)
#pragma unroll
        for (int s = 0; s < 2; ++s)
          acc[mt][nt] = wmma_bf16(afrag[mt][s], bfrag[nt][s], acc[mt][nt]);
    __syncthreads();
  }
  // Epilogue: bias is reused across mt/r; only 2 distinct columns per thread.
  float bv[2];
#pragma unroll
  for (int nt = 0; nt < 2; ++nt) bv[nt] = bias[n0 + wn * 32 + nt * 16 + lr];
#pragma unroll
  for (int mt = 0; mt < 2; ++mt)
#pragma unroll
    for (int nt = 0; nt < 2; ++nt) {
      const int gn = n0 + wn * 32 + nt * 16 + lr;
#pragma unroll
      for (int r = 0; r < 8; ++r) {
        const int gm = m0 + wm * 32 + mt * 16 + r + half * 8;
        out[(size_t)gm * DIM + gn] = acc[mt][nt][r] + bv[nt];
      }
    }
}

extern "C" void kernel_launch(void* const* d_in, const int* in_sizes, int n_in,
                              void* d_out, int out_size, void* d_ws, size_t ws_size,
                              hipStream_t stream) {
  const float* x     = (const float*)d_in[0];
  const float* w_qkv = (const float*)d_in[1];
  const float* w_out = (const float*)d_in[2];
  const float* b_out = (const float*)d_in[3];
  float* out = (float*)d_out;

  const size_t per = (size_t)BATCH * HEADS * SEQ * HEAD_DIM;  // 6291456 elems
  bf16_t* qw  = (bf16_t*)d_ws;
  bf16_t* kw  = qw + per;
  bf16_t* vtw = kw + per;
  bf16_t* ow  = vtw + per;  // [8192,768] bf16; total ws: 4*per*2 = 50331648 B

  qkv_gemm_kernel<<<dim3(QKV_N / TN, MROWS / TM), 256, 0, stream>>>(x, w_qkv, qw, kw, vtw);
  attn_kernel<<<dim3(SEQ / 64, BATCH * HEADS), 128, 0, stream>>>(qw, kw, vtw, ow);
  out_gemm_kernel<<<dim3(DIM / TN, MROWS / TM), 256, 0, stream>>>(ow, w_out, b_out, out);
}
